// BiDBlayerCrystalGraphConvNet_56667798504177
// MI455X (gfx1250) — matrix operand order repaired and checked by reading, not compiled, not run on hardware
//
#include <hip/hip_runtime.h>

typedef __bf16  v16bf __attribute__((ext_vector_type(16)));
typedef __bf16  v8bf  __attribute__((ext_vector_type(8)));
typedef float   v8f   __attribute__((ext_vector_type(8)));

#define N_ATOMS 131072
#define M_NBR   12
#define NM_ROWS (N_ATOMS * M_NBR)   // 1572864
#define AFEA    64
#define C2      128                 // 2*AFEA
#define NBRF    41
#define FIN     169                 // 2*AFEA + NBRF
#define KPAD    192                 // FIN padded to 6 chunks of 32
#define B_CRYS  2048
#define HFEA    128
#define RPB     128                 // rows per block in GEMM kernels

// Native converts: gfx1250 has bf16 cvt hardware; let clang pick v_cvt/pk forms.
static __device__ __forceinline__ __bf16 f2bf(float f) {
    return (__bf16)f;
}
static __device__ __forceinline__ float bf2f(__bf16 h) {
    unsigned short u = __builtin_bit_cast(unsigned short, h);
    unsigned x = ((unsigned)u) << 16;
    return __builtin_bit_cast(float, x);
}
static __device__ __forceinline__ float softplusf(float x) {
    return x > 20.f ? x : log1pf(__expf(x));
}
static __device__ __forceinline__ float sigmoidf(float x) {
    return 1.f / (1.f + __expf(-x));
}
// Combine two contiguous 8-element LDS loads into one v16bf WMMA operand
static __device__ __forceinline__ v16bf cat16(v8bf lo, v8bf hi) {
    return __builtin_shufflevector(lo, hi, 0,1,2,3,4,5,6,7,8,9,10,11,12,13,14,15);
}

// ---------------------------------------------------------------------------
// Embedding GEMM: (N,92) @ (92,64) + b  ->  afea (N,64) f32.  K padded to 96.
// Block: 256 thr = 8 waves; wave does 16 rows x 64 cols (4 WMMA col tiles).
// ---------------------------------------------------------------------------
__global__ __launch_bounds__(256) void embed_gemm(
    const float* __restrict__ atom, const float* __restrict__ W,
    const float* __restrict__ bias, float* __restrict__ afea)
{
    __shared__ __align__(16) __bf16 sA[RPB][96];        // 24 KB, row stride 192B
    __shared__ __align__(32) __bf16 sBs[3][4][32][16];  // 12 KB, swizzled B frags
    const int tid = threadIdx.x, lane = tid & 31, wave = tid >> 5;
    const int rowBase = blockIdx.x * RPB;

    for (int e = tid; e < RPB * 96; e += 256) {
        int r = e / 96, k = e - r * 96;
        sA[r][k] = f2bf(k < 92 ? atom[(size_t)(rowBase + r) * 92 + k] : 0.f);
    }
    // Pre-swizzle W into B-fragment layout: sBs[kc][t][lane][i] = W[kc*32 + (lane>>4)*16 + i][t*16 + (lane&15)]
    for (int e = tid; e < 96 * AFEA; e += 256) {
        int k = e >> 6, c = e & 63;
        int kc = k >> 5, kr = k & 31;
        int t = c >> 4;
        int bl = (kr >> 4) * 16 + (c & 15);
        int i  = kr & 15;
        sBs[kc][t][bl][i] = f2bf(k < 92 ? W[k * AFEA + c] : 0.f);
    }
    __syncthreads();

    const v8f z8 = {0.f,0.f,0.f,0.f,0.f,0.f,0.f,0.f};
    v8f acc[4] = {z8, z8, z8, z8};
    const int aRow  = wave * 16 + (lane & 15);
    const int kHalf = lane >> 4;

    #pragma unroll
    for (int kc = 0; kc < 3; ++kc) {
        const v8bf* ap = (const v8bf*)&sA[aRow][kc * 32 + kHalf * 16];
        v16bf a = cat16(ap[0], ap[1]);      // K = kb..kb+15
        #pragma unroll
        for (int t = 0; t < 4; ++t) {
            v16bf b = *(const v16bf*)&sBs[kc][t][lane][0];
            acc[t] = __builtin_amdgcn_wmma_f32_16x16x32_bf16(
                false, a, false, b, (short)0, acc[t], false, false);
        }
    }
    const int r0 = rowBase + wave * 16 + (lane >> 4) * 8;
    #pragma unroll
    for (int t = 0; t < 4; ++t) {
        const int col = t * 16 + (lane & 15);
        const float bv = bias[col];
        #pragma unroll
        for (int j = 0; j < 8; ++j)
            afea[(size_t)(r0 + j) * AFEA + col] = acc[t][j] + bv;
    }
}

// ---------------------------------------------------------------------------
// Conv GEMM: rows of total=[self||afea[idx]||nbr] (NM x 169, padded 192)
// times Wc (169x128) + bc.  Stores gated (NM x 128) as bf16 and accumulates
// per-column sum / sumsq for BN1 batch statistics.
// Block: 256 thr = 8 waves; wave: 16 rows x 128 cols (8 col tiles, 6 k-chunks).
// ---------------------------------------------------------------------------
__global__ __launch_bounds__(256) void conv_gemm(
    const float* __restrict__ afea, const float* __restrict__ nbr,
    const int* __restrict__ idx, const float* __restrict__ W,
    const float* __restrict__ bias, __bf16* __restrict__ gated,
    float* __restrict__ sum1, float* __restrict__ sumsq1)
{
    __shared__ __align__(16) __bf16 sA[RPB][KPAD];      // 49152 B, row stride 384B
    __shared__ __align__(32) __bf16 sBs[8][32][16];     // 8192 B, swizzled B frags
    __shared__ float sSum[C2];
    __shared__ float sSq[C2];

    const int tid = threadIdx.x, lane = tid & 31, wave = tid >> 5;
    const int rowBase = blockIdx.x * RPB;

    if (tid < C2) { sSum[tid] = 0.f; sSq[tid] = 0.f; }

    // Stage A panel: gather + concat + pad + bf16 convert
    for (int e = tid; e < RPB * KPAD; e += 256) {
        int r = e / KPAD, k = e - r * KPAD;
        int grow = rowBase + r;
        int n = grow / M_NBR;
        float v;
        if (k < AFEA)           v = afea[(size_t)n * AFEA + k];
        else if (k < 2 * AFEA)  v = afea[(size_t)idx[grow] * AFEA + (k - AFEA)];
        else if (k < FIN)       v = nbr[(size_t)grow * NBRF + (k - 2 * AFEA)];
        else                    v = 0.f;
        sA[r][k] = f2bf(v);
    }
    __syncthreads();

    const v8f z8 = {0.f,0.f,0.f,0.f,0.f,0.f,0.f,0.f};
    v8f acc[8] = {z8, z8, z8, z8, z8, z8, z8, z8};
    const int aRow  = wave * 16 + (lane & 15);
    const int kHalf = lane >> 4;

    #pragma unroll
    for (int kc = 0; kc < KPAD / 32; ++kc) {
        __syncthreads();                       // protect sBs reuse across chunks
        // Pre-swizzle W chunk into B-fragment layout:
        // sBs[t][bl][i] = W[kc*32 + (bl>>4)*16 + i][t*16 + (bl&15)]
        for (int e = tid; e < 32 * C2; e += 256) {
            int kr = e >> 7, c = e & 127;
            int gk = kc * 32 + kr;
            int t  = c >> 4;
            int bl = (kr >> 4) * 16 + (c & 15);
            int i  = kr & 15;
            sBs[t][bl][i] = f2bf(gk < FIN ? W[gk * C2 + c] : 0.f);
        }
        __syncthreads();

        const v8bf* ap = (const v8bf*)&sA[aRow][kc * 32 + kHalf * 16];
        v16bf a = cat16(ap[0], ap[1]);         // two ds_load_b128
        #pragma unroll
        for (int t = 0; t < 8; ++t) {
            v16bf b = *(const v16bf*)&sBs[t][lane][0];
            acc[t] = __builtin_amdgcn_wmma_f32_16x16x32_bf16(
                false, a, false, b, (short)0, acc[t], false, false);
        }
    }

    // Epilogue: bias + BN1 partial stats (LDS atomics)
    #pragma unroll
    for (int t = 0; t < 8; ++t) {
        const float bv = bias[t * 16 + (lane & 15)];
        float s = 0.f, q = 0.f;
        #pragma unroll
        for (int j = 0; j < 8; ++j) {
            float v = acc[t][j] + bv;
            acc[t][j] = v;
            s += v; q += v * v;
        }
        atomicAdd(&sSum[t * 16 + (lane & 15)], s);
        atomicAdd(&sSq [t * 16 + (lane & 15)], q);
    }
    __syncthreads();

    // Reuse sA as bf16 C-staging (128x128 = 32 KB) for coalesced stores
    __bf16 (*sC)[C2] = (__bf16(*)[C2]) &sA[0][0];
    const int r0 = wave * 16 + (lane >> 4) * 8;
    #pragma unroll
    for (int t = 0; t < 8; ++t) {
        const int col = t * 16 + (lane & 15);
        #pragma unroll
        for (int j = 0; j < 8; ++j)
            sC[r0 + j][col] = f2bf(acc[t][j]);
    }
    __syncthreads();

    if (tid < C2) {
        atomicAdd(&sum1[tid],   sSum[tid]);
        atomicAdd(&sumsq1[tid], sSq[tid]);
    }
    const uint4* src = (const uint4*)&sC[0][0];
    uint4* dst = (uint4*)gated + (size_t)rowBase * (C2 / 8);
    for (int v = tid; v < RPB * C2 / 8; v += 256) dst[v] = src[v];
}

// ---------------------------------------------------------------------------
__global__ void finalize_stats(const float* __restrict__ sum,
                               const float* __restrict__ sumsq,
                               float* __restrict__ mean, float* __restrict__ inv,
                               int n, float count)
{
    int i = threadIdx.x;
    if (i < n) {
        float mu  = sum[i] / count;
        float var = sumsq[i] / count - mu * mu;
        mean[i] = mu;
        inv[i]  = rsqrtf(var + 1e-5f);
    }
}

// BN1 apply + sigmoid*softplus gate + sum over M neighbors; BN2 stats.
// Block 256 = 4 atoms x 64 channels.
__global__ __launch_bounds__(256) void gate_sum(
    const __bf16* __restrict__ gated,
    const float* __restrict__ mean1, const float* __restrict__ inv1,
    const float* __restrict__ g1,    const float* __restrict__ b1,
    float* __restrict__ nbrS, float* __restrict__ sum2, float* __restrict__ sumsq2)
{
    __shared__ float sSum[AFEA], sSq[AFEA];
    const int tid = threadIdx.x;
    if (tid < AFEA) { sSum[tid] = 0.f; sSq[tid] = 0.f; }
    __syncthreads();

    const int g = tid >> 6, c = tid & 63;
    const int n = blockIdx.x * 4 + g;
    const float mA = mean1[c],      iA = inv1[c],      gA = g1[c],      bA = b1[c];
    const float mB = mean1[c + 64], iB = inv1[c + 64], gB = g1[c + 64], bB = b1[c + 64];
    float s = 0.f;
    for (int m = 0; m < M_NBR; ++m) {
        size_t row = (size_t)n * M_NBR + m;
        float xf = (bf2f(gated[row * C2 + c])      - mA) * iA * gA + bA;
        float xc = (bf2f(gated[row * C2 + c + 64]) - mB) * iB * gB + bB;
        s += sigmoidf(xf) * softplusf(xc);
    }
    nbrS[(size_t)n * AFEA + c] = s;
    atomicAdd(&sSum[c], s);
    atomicAdd(&sSq[c],  s * s);
    __syncthreads();
    if (tid < AFEA) {
        atomicAdd(&sum2[tid],   sSum[tid]);
        atomicAdd(&sumsq2[tid], sSq[tid]);
    }
}

// BN2 apply + residual + softplus
__global__ void update_afea(
    const float* __restrict__ afea_in, const float* __restrict__ nbrS,
    const float* __restrict__ mean2, const float* __restrict__ inv2,
    const float* __restrict__ g2, const float* __restrict__ b2,
    float* __restrict__ afea_out)
{
    int i = blockIdx.x * blockDim.x + threadIdx.x;
    int c = i & 63;
    float v = (nbrS[i] - mean2[c]) * inv2[c] * g2[c] + b2[c];
    afea_out[i] = softplusf(afea_in[i] + v);
}

// Segment-sum pooling via global atomics
__global__ void pool_sum(const float* __restrict__ afea, const int* __restrict__ crys,
                         float* __restrict__ pooled, float* __restrict__ cnt)
{
    int i = blockIdx.x * 256 + threadIdx.x;   // N*64 total
    int n = i >> 6, c = i & 63;
    int b = crys[n];
    atomicAdd(&pooled[(size_t)b * AFEA + c], afea[i]);
    if (c == 0) atomicAdd(&cnt[b], 1.f);
}

__global__ __launch_bounds__(128) void head_fc(
    const float* __restrict__ pooled, const float* __restrict__ cnt,
    const float* __restrict__ W, const float* __restrict__ b, float* __restrict__ out)
{
    __shared__ float x[AFEA];
    const int bid = blockIdx.x, j = threadIdx.x;
    const float c = fmaxf(cnt[bid], 1.f);
    if (j < AFEA) x[j] = pooled[(size_t)bid * AFEA + j] / c;
    __syncthreads();
    float s = b[j];
    for (int k = 0; k < AFEA; ++k) s += x[k] * W[k * HFEA + j];
    out[(size_t)bid * HFEA + j] = softplusf(s);
}

__global__ __launch_bounds__(128) void head_f1(
    const float* __restrict__ in, const float* __restrict__ W,
    const float* __restrict__ b, float* __restrict__ out)
{
    __shared__ float x[HFEA];
    const int bid = blockIdx.x, j = threadIdx.x;
    x[j] = in[(size_t)bid * HFEA + j];
    __syncthreads();
    float s = b[j];
    for (int k = 0; k < HFEA; ++k) s += x[k] * W[k * HFEA + j];
    out[(size_t)bid * HFEA + j] = fmaxf(s, 0.f);
}

__global__ __launch_bounds__(128) void head_out(
    const float* __restrict__ in, const float* __restrict__ W,
    const float* __restrict__ b, float* __restrict__ out)
{
    __shared__ float red[HFEA];
    const int bid = blockIdx.x, j = threadIdx.x;
    red[j] = in[(size_t)bid * HFEA + j] * W[j];
    __syncthreads();
    for (int s = 64; s > 0; s >>= 1) {
        if (j < s) red[j] += red[j + s];
        __syncthreads();
    }
    if (j == 0) out[bid] = red[0] + b[0];
}

__global__ void zero_f(float* p, int n)
{
    int i = blockIdx.x * 256 + threadIdx.x;
    if (i < n) p[i] = 0.f;
}

// ---------------------------------------------------------------------------
extern "C" void kernel_launch(void* const* d_in, const int* in_sizes, int n_in,
                              void* d_out, int out_size, void* d_ws, size_t ws_size,
                              hipStream_t stream)
{
    (void)in_sizes; (void)n_in; (void)out_size; (void)ws_size;
    const float* atom  = (const float*)d_in[0];
    const float* nbr   = (const float*)d_in[1];
    const int*   idx   = (const int*)  d_in[2];
    const int*   crys  = (const int*)  d_in[3];
    /* d_in[4] mono_bg: unused by reference */
    const float* W_emb = (const float*)d_in[5];
    const float* b_emb = (const float*)d_in[6];
    const float* Wc    = (const float*)d_in[7];
    const float* bc    = (const float*)d_in[8];
    const float* g1    = (const float*)d_in[9];
    const float* b1    = (const float*)d_in[10];
    const float* g2    = (const float*)d_in[11];
    const float* b2    = (const float*)d_in[12];
    const float* W_fc  = (const float*)d_in[13];
    const float* b_fc  = (const float*)d_in[14];
    const float* W_f1  = (const float*)d_in[15];
    const float* b_f1  = (const float*)d_in[16];
    const float* W_out = (const float*)d_in[17];
    const float* b_out = (const float*)d_in[18];

    // ---- workspace carve (aligned 256B) ----
    char* w = (char*)d_ws;
    auto carve = [&](size_t bytes) -> char* {
        char* p = w;
        w += (bytes + 255) & ~(size_t)255;
        return p;
    };
    float* afeaA  = (float*)carve((size_t)N_ATOMS * AFEA * 4);
    float* afeaB  = (float*)carve((size_t)N_ATOMS * AFEA * 4);
    float* nbrS   = (float*)carve((size_t)N_ATOMS * AFEA * 4);
    float* stats  = (float*)carve(1024 * 4);
    float* sum1   = stats;        float* sumsq1 = stats + 128;
    float* mean1  = stats + 256;  float* inv1   = stats + 384;
    float* sum2   = stats + 512;  float* sumsq2 = stats + 576;
    float* mean2  = stats + 640;  float* inv2   = stats + 704;
    float* pooled = (float*)carve((size_t)B_CRYS * AFEA * 4 + B_CRYS * 4);
    float* cnt    = pooled + (size_t)B_CRYS * AFEA;
    float* crysF  = (float*)carve((size_t)B_CRYS * HFEA * 4);
    float* fusedF = (float*)carve((size_t)B_CRYS * HFEA * 4);
    __bf16* gated = (__bf16*)carve((size_t)NM_ROWS * C2 * 2);

    // ---- embedding ----
    embed_gemm<<<N_ATOMS / RPB, 256, 0, stream>>>(atom, W_emb, b_emb, afeaA);

    // ---- 3 conv layers ----
    float* cur = afeaA;
    float* nxt = afeaB;
    for (int l = 0; l < 3; ++l) {
        zero_f<<<3, 256, 0, stream>>>(stats, 768);
        conv_gemm<<<NM_ROWS / RPB, 256, 0, stream>>>(
            cur, nbr, idx, Wc + (size_t)l * FIN * C2, bc + l * C2,
            gated, sum1, sumsq1);
        finalize_stats<<<1, 128, 0, stream>>>(sum1, sumsq1, mean1, inv1,
                                              C2, (float)NM_ROWS);
        gate_sum<<<N_ATOMS / 4, 256, 0, stream>>>(
            gated, mean1, inv1, g1 + l * C2, b1 + l * C2, nbrS, sum2, sumsq2);
        finalize_stats<<<1, 128, 0, stream>>>(sum2, sumsq2, mean2, inv2,
                                              AFEA, (float)N_ATOMS);
        update_afea<<<(N_ATOMS * AFEA) / 256, 256, 0, stream>>>(
            cur, nbrS, mean2, inv2, g2 + l * AFEA, b2 + l * AFEA, nxt);
        float* t = cur; cur = nxt; nxt = t;
    }

    // ---- segment-mean pooling + head MLP ----
    zero_f<<<(B_CRYS * AFEA + B_CRYS + 255) / 256, 256, 0, stream>>>(
        pooled, B_CRYS * AFEA + B_CRYS);
    pool_sum<<<(N_ATOMS * AFEA) / 256, 256, 0, stream>>>(cur, crys, pooled, cnt);
    head_fc <<<B_CRYS, 128, 0, stream>>>(pooled, cnt, W_fc, b_fc, crysF);
    head_f1 <<<B_CRYS, 128, 0, stream>>>(crysF, W_f1, b_f1, fusedF);
    head_out<<<B_CRYS, 128, 0, stream>>>(fusedF, W_out, b_out, (float*)d_out);
}
